// Attention_5677946766100
// MI455X (gfx1250) — matrix operand order repaired and checked by reading, not compiled, not run on hardware
//
#include <hip/hip_runtime.h>

#define BQ    128   // query rows per block
#define NW    8     // waves per block (wave32)
#define KC    128   // key chunk staged in LDS
#define KI    64    // inner key tile (softmax amortization width)
#define DHEAD 64
#define SLEN  2048
#define NBH   64    // B*H

typedef __attribute__((ext_vector_type(16))) __bf16 v16bf;
typedef __attribute__((ext_vector_type(8)))  float  v8f;

union FragBF {
    v16bf v;
    unsigned long long u64[4];
    unsigned short u16[16];
};

// f32 -> bf16 round-to-nearest-even (for Q/K/V staging)
__device__ __forceinline__ unsigned short f2bf(float x) {
    unsigned int u = __float_as_uint(x);
    u += 0x7FFFu + ((u >> 16) & 1u);
    return (unsigned short)(u >> 16);
}
// f32 -> bf16 truncation (for P in [0,1]; fuses into d16_hi stores)
__device__ __forceinline__ unsigned short f2bf_trunc(float x) {
    return (unsigned short)(__float_as_uint(x) >> 16);
}

// Cross-lane move via DPP (stays in VALU, no LDS traffic).
template <int CTRL>
__device__ __forceinline__ float dpp_mov_f32(float x) {
    int r = __builtin_amdgcn_update_dpp(0, __float_as_int(x), CTRL,
                                        /*row_mask=*/0xF, /*bank_mask=*/0xF,
                                        /*bound_ctrl=*/true);
    return __int_as_float(r);
}

// Butterfly reductions over a 16-lane DPP row (both halves independent).
// xor masks {1,2,7,15} span the 4-bit lane space -> full 16-lane reduction.
__device__ __forceinline__ float rowmax16(float x) {
    x = fmaxf(x, dpp_mov_f32<0xB1>(x));    // quad_perm(1,0,3,2)  = xor 1
    x = fmaxf(x, dpp_mov_f32<0x4E>(x));    // quad_perm(2,3,0,1)  = xor 2
    x = fmaxf(x, dpp_mov_f32<0x141>(x));   // row_half_mirror     = xor 7
    x = fmaxf(x, dpp_mov_f32<0x140>(x));   // row_mirror          = xor 15
    return x;
}
__device__ __forceinline__ float rowsum16(float x) {
    x += dpp_mov_f32<0xB1>(x);
    x += dpp_mov_f32<0x4E>(x);
    x += dpp_mov_f32<0x141>(x);
    x += dpp_mov_f32<0x140>(x);
    return x;
}

__global__ __launch_bounds__(256)
void Attention_5677946766100_kernel(const float* __restrict__ Q,
                                    const float* __restrict__ K,
                                    const float* __restrict__ V,
                                    float* __restrict__ O) {
    // LDS staging. Padding chosen so 16-lane fragment gathers hit distinct banks.
    __shared__ unsigned short sK [KC][DHEAD + 4];   // K chunk, [key][d]
    __shared__ unsigned short sVt[DHEAD][KC + 4];   // V chunk transposed, [d][key]
    __shared__ unsigned short sP [NW][16][KI + 4];  // per-wave P re-layout scratch

    const int tid   = threadIdx.x;
    const int wave  = tid >> 5;
    const int lane  = tid & 31;
    const int lhalf = lane >> 4;     // which 16-lane half
    const int l16   = lane & 15;

    const int    bh    = blockIdx.y;
    const size_t base  = (size_t)bh * SLEN * DHEAD;
    const int    qbase = blockIdx.x * BQ + wave * 16;

    // ---- Q as two 16x32 bf16 A-fragments (d = 0..31, 32..63), pre-scaled by 1/sqrt(d) ----
    // A layout: M = lane&15, half h -> k = h + 16*lhalf
    FragBF qa0, qa1;
    {
        const float* qrow = Q + base + (size_t)(qbase + l16) * DHEAD;
#pragma unroll
        for (int h = 0; h < 16; ++h) {
            int d = h + (lhalf << 4);
            qa0.u16[h] = f2bf(qrow[d] * 0.125f);
            qa1.u16[h] = f2bf(qrow[d + 32] * 0.125f);
        }
    }

    float mrow[8], lrow[8];
    v8f   oacc[4];
#pragma unroll
    for (int r = 0; r < 8; ++r) { mrow[r] = -1e30f; lrow[r] = 0.0f; }
#pragma unroll
    for (int nt = 0; nt < 4; ++nt) oacc[nt] = (v8f){};

    for (int kc = 0; kc < SLEN; kc += KC) {
        // ---- stage 128 keys of K and V (f32 -> bf16), V transposed ----
        const float4* kg = (const float4*)(K + base + (size_t)kc * DHEAD);
        const float4* vg = (const float4*)(V + base + (size_t)kc * DHEAD);
#pragma unroll
        for (int i = 0; i < 8; ++i) {
            int    idx = tid + i * 256;      // float4 index in chunk
            float4 kv  = kg[idx];
            float4 vv  = vg[idx];
            int e   = idx << 2;
            int key = e >> 6;                // / DHEAD
            int d   = e & 63;
            sK[key][d + 0] = f2bf(kv.x);
            sK[key][d + 1] = f2bf(kv.y);
            sK[key][d + 2] = f2bf(kv.z);
            sK[key][d + 3] = f2bf(kv.w);
            sVt[d + 0][key] = f2bf(vv.x);
            sVt[d + 1][key] = f2bf(vv.y);
            sVt[d + 2][key] = f2bf(vv.z);
            sVt[d + 3][key] = f2bf(vv.w);
        }
        if (kc + KC < SLEN) {
            __builtin_prefetch(K + base + (size_t)(kc + KC) * DHEAD, 0, 1);
            __builtin_prefetch(V + base + (size_t)(kc + KC) * DHEAD, 0, 1);
        }
        __syncthreads();

        for (int ki = 0; ki < KC; ki += KI) {
            // ---- scores: S = Q * K^T for 4 key tiles of 16 ----
            // B layout (32x16, K=d): N = key = l16, half h -> d = 32*dh + 16*lhalf + h
            v8f c[4];
#pragma unroll
            for (int t = 0; t < 4; ++t) {
                int key = ki + t * 16 + l16;
                FragBF b0, b1;
                const unsigned long long* p0 = (const unsigned long long*)&sK[key][lhalf * 16];
                const unsigned long long* p1 = (const unsigned long long*)&sK[key][32 + lhalf * 16];
#pragma unroll
                for (int j = 0; j < 4; ++j) { b0.u64[j] = p0[j]; b1.u64[j] = p1[j]; }
                c[t] = (v8f){};
                c[t] = __builtin_amdgcn_wmma_f32_16x16x32_bf16(false, qa0.v, false, b0.v, (short)0, c[t], false, false);
                c[t] = __builtin_amdgcn_wmma_f32_16x16x32_bf16(false, qa1.v, false, b1.v, (short)0, c[t], false, false);
            }

            // ---- online softmax over 64 keys (DPP row reductions, no LDS) ----
            // C layout: VGPR r -> row M = r + 8*lhalf, 16 N-values across a 16-lane half
#pragma unroll
            for (int r = 0; r < 8; ++r) {
                float s0 = c[0][r], s1 = c[1][r], s2 = c[2][r], s3 = c[3][r];
                float mx   = rowmax16(fmaxf(fmaxf(s0, s1), fmaxf(s2, s3)));
                float mnew = fmaxf(mrow[r], mx);
                float p0 = __expf(s0 - mnew);
                float p1 = __expf(s1 - mnew);
                float p2 = __expf(s2 - mnew);
                float p3 = __expf(s3 - mnew);
                float rs = rowsum16((p0 + p1) + (p2 + p3));
                float corr = __expf(mrow[r] - mnew);
                mrow[r] = mnew;
                lrow[r] = lrow[r] * corr + rs;
                oacc[0][r] *= corr; oacc[1][r] *= corr;
                oacc[2][r] *= corr; oacc[3][r] *= corr;
                int m = r + (lhalf << 3);
                sP[wave][m][l16]      = f2bf_trunc(p0);
                sP[wave][m][16 + l16] = f2bf_trunc(p1);
                sP[wave][m][32 + l16] = f2bf_trunc(p2);
                sP[wave][m][48 + l16] = f2bf_trunc(p3);
            }

            // ---- re-layout P (C-layout) -> two 16x32 bf16 A-fragments via per-wave LDS ----
            FragBF pa0, pa1;
            {
                const unsigned long long* p = (const unsigned long long*)&sP[wave][l16][0];
                pa0.u64[0] = p[lhalf * 4 + 0];   // k = 16*lhalf      .. +15
                pa0.u64[1] = p[lhalf * 4 + 1];
                pa0.u64[2] = p[lhalf * 4 + 2];
                pa0.u64[3] = p[lhalf * 4 + 3];
                pa1.u64[0] = p[lhalf * 4 + 8];   // k = 32 + 16*lhalf .. +15
                pa1.u64[1] = p[lhalf * 4 + 9];
                pa1.u64[2] = p[lhalf * 4 + 10];
                pa1.u64[3] = p[lhalf * 4 + 11];
            }

            // ---- O += P * V, four 16-wide d tiles x two K=32 fragments ----
            // B layout: N = d = 16*nt + l16, half h -> key = kb + 16*lhalf + h (contiguous in sVt)
#pragma unroll
            for (int nt = 0; nt < 4; ++nt) {
                FragBF vb0, vb1;
                const unsigned long long* q0 =
                    (const unsigned long long*)&sVt[nt * 16 + l16][ki + lhalf * 16];
                const unsigned long long* q1 =
                    (const unsigned long long*)&sVt[nt * 16 + l16][ki + 32 + lhalf * 16];
#pragma unroll
                for (int j = 0; j < 4; ++j) { vb0.u64[j] = q0[j]; vb1.u64[j] = q1[j]; }
                oacc[nt] = __builtin_amdgcn_wmma_f32_16x16x32_bf16(
                    false, pa0.v, false, vb0.v, (short)0, oacc[nt], false, false);
                oacc[nt] = __builtin_amdgcn_wmma_f32_16x16x32_bf16(
                    false, pa1.v, false, vb1.v, (short)0, oacc[nt], false, false);
            }
        }
        __syncthreads();
    }

    // ---- epilogue: O / l, write fp32 ----
#pragma unroll
    for (int r = 0; r < 8; ++r) {
        float inv = 1.0f / lrow[r];
        int   m   = qbase + r + (lhalf << 3);
        float* orow = O + base + (size_t)m * DHEAD + l16;
        orow[0]  = oacc[0][r] * inv;
        orow[16] = oacc[1][r] * inv;
        orow[32] = oacc[2][r] * inv;
        orow[48] = oacc[3][r] * inv;
    }
}

extern "C" void kernel_launch(void* const* d_in, const int* in_sizes, int n_in,
                              void* d_out, int out_size, void* d_ws, size_t ws_size,
                              hipStream_t stream) {
    (void)in_sizes; (void)n_in; (void)out_size; (void)d_ws; (void)ws_size;
    const float* Q = (const float*)d_in[0];
    const float* K = (const float*)d_in[1];
    const float* V = (const float*)d_in[2];
    float* Out = (float*)d_out;
    dim3 grid(SLEN / BQ, NBH);   // 16 x 64 blocks
    dim3 block(256);             // 8 wave32s
    Attention_5677946766100_kernel<<<grid, block, 0, stream>>>(Q, K, V, Out);
}
